// VGG16_FPN_87488483820370
// MI455X (gfx1250) — compile-verified, matrix operand order checked by reading, not run on hardware
//
#include <hip/hip_runtime.h>
#include <hip/hip_bf16.h>
#include <stdint.h>

#define BB 8
#define CC 512
#define HH 48
#define WWW 48
#define NN (HH * WWW)   // 2304
#define TOPK 3
#define MAXDEG 8
#define LN_EPS 1e-5f

typedef __attribute__((ext_vector_type(16))) _Float16 v16h;
typedef __attribute__((ext_vector_type(8)))  _Float16 v8h;
typedef __attribute__((ext_vector_type(8)))  float    v8f;
typedef __attribute__((ext_vector_type(4)))  uint32_t u32x4;
typedef __attribute__((ext_vector_type(8)))  uint32_t u32x8;

// ---------------------------------------------------------------------------
// WMMA helper (probe-verified signature for gfx1250)
// ---------------------------------------------------------------------------
__device__ inline v8f wmma_f16(v16h a, v16h b, v8f c) {
  return __builtin_amdgcn_wmma_f32_16x16x32_f16(
      /*neg_a=*/false, a, /*neg_b=*/false, b,
      /*c_mod=*/(short)0, c, /*reuse_a=*/false, /*reuse_b=*/false);
}

// Per-lane A/B fragment loader for 16x16x32 f16 WMMA.
// Lane l: row/col = l&15, khalf = l>>4.  Elements cover K = khalf*8 + [0..7]
// and K = 16 + khalf*8 + [0..7]  (two 16-byte LDS chunks -> ds_load_b128).
__device__ inline v16h ld_frag(const _Float16* rowbase32, int khalf) {
  v8h lo = *(const v8h*)(rowbase32 + khalf * 8);
  v8h hi = *(const v8h*)(rowbase32 + 16 + khalf * 8);
  return __builtin_shufflevector(lo, hi, 0,1,2,3,4,5,6,7,8,9,10,11,12,13,14,15);
}

// LDS byte offset (low 32 bits of the generic address == DS address)
__device__ inline uint32_t lds_off(const void* p) {
  return (uint32_t)(uintptr_t)p;
}

// Async global->LDS 16-byte copy (per-lane), tracked by ASYNCcnt.
__device__ inline void async_copy_b128(uint32_t ldsAddr, const void* gptr) {
  asm volatile("global_load_async_to_lds_b128 %0, %1, off"
               :: "v"(ldsAddr), "v"((uint64_t)(uintptr_t)gptr)
               : "memory");
}
__device__ inline void wait_asynccnt0() {
  asm volatile("s_wait_asynccnt 0x0" ::: "memory");
}

__device__ inline uint32_t rfl(uint32_t x) {
  return (uint32_t)__builtin_amdgcn_readfirstlane((int)x);
}

// Issue one TDM tensor_load_to_lds for a 2D tile of `rows` x `cols` f16
// elements, row stride `stride` (elements), contiguous into LDS.
// Must be executed by exactly one wave (wave-uniform call site).
__device__ inline void tdm_load_tile_f16(uint32_t ldsAddr, const void* gptr,
                                         uint32_t cols, uint32_t rows,
                                         uint32_t stride) {
  uint64_t ga = (uint64_t)(uintptr_t)gptr;
  u32x4 g0;
  g0.x = rfl(1u);                                   // count=1, user mode
  g0.y = rfl(ldsAddr);                              // lds_addr
  g0.z = rfl((uint32_t)ga);                         // global_addr[31:0]
  g0.w = rfl(((uint32_t)(ga >> 32) & 0x01ffffffu) | (2u << 30)); // [56:32]|type=2
  u32x8 g1;
  g1.s0 = rfl(1u << 16);                            // data_size=1 (2 bytes)
  g1.s1 = rfl((cols & 0xffffu) << 16);              // tensor_dim0[15:0]
  g1.s2 = rfl((cols >> 16) | ((rows & 0xffffu) << 16)); // td0[31:16] | td1[15:0]
  g1.s3 = rfl((rows >> 16) | (cols << 16));         // td1[31:16] | tile_dim0
  g1.s4 = rfl(rows & 0xffffu);                      // tile_dim1 (tile_dim2=0)
  g1.s5 = rfl(stride);                              // tensor_dim0_stride[31:0]
  g1.s6 = rfl(0u);                                  // stride[47:32] | td1_stride lo
  g1.s7 = rfl(0u);
  asm volatile("tensor_load_to_lds %0, %1" :: "s"(g0), "s"(g1) : "memory");
}

// ---------------------------------------------------------------------------
// 0) Convert 8 weight matrices to f16, transposed (wt[n][k] = w[k][n]).
// ---------------------------------------------------------------------------
__global__ void cvt_weights_kernel(const float* w0, const float* w1,
                                   const float* w2, const float* w3,
                                   const float* w4, const float* w5,
                                   const float* w6, const float* w7,
                                   _Float16* wt) {
  const float* w;
  switch (blockIdx.y) {
    case 0: w = w0; break;  case 1: w = w1; break;
    case 2: w = w2; break;  case 3: w = w3; break;
    case 4: w = w4; break;  case 5: w = w5; break;
    case 6: w = w6; break;  default: w = w7; break;
  }
  int e = blockIdx.x * blockDim.x + threadIdx.x;     // [0, C*C)
  int n = e / CC, k = e % CC;
  wt[(size_t)blockIdx.y * CC * CC + (size_t)n * CC + k] = (_Float16)w[(size_t)k * CC + n];
}

// 1) Elementwise f32 -> f16
__global__ void cvt_f16_kernel(const float* x, _Float16* y, int total) {
  int i = blockIdx.x * blockDim.x + threadIdx.x;
  if (i < total) y[i] = (_Float16)x[i];
}

// 2) Squared norms of every node row
__global__ void sqnorm_kernel(const float* x, float* sq) {
  __shared__ float red[128];
  const float* row = x + (size_t)blockIdx.x * CC;
  float s = 0.f;
  for (int c = threadIdx.x; c < CC; c += 128) { float v = row[c]; s += v * v; }
  red[threadIdx.x] = s;
  __syncthreads();
  for (int st = 64; st > 0; st >>= 1) {
    if (threadIdx.x < st) red[threadIdx.x] += red[threadIdx.x + st];
    __syncthreads();
  }
  if (threadIdx.x == 0) sq[blockIdx.x] = red[0];
}

// 3) Spatial 3-NN on the 48x48 grid (lexicographic (d2, index) tie-break)
__global__ void spatial_knn_kernel(int* sidx) {
  int i = blockIdx.x * blockDim.x + threadIdx.x;
  if (i >= NN) return;
  int yi = i / WWW, xi = i % WWW;
  int bd[TOPK]; int bi[TOPK];
#pragma unroll
  for (int t = 0; t < TOPK; ++t) { bd[t] = 0x7fffffff; bi[t] = -1; }
  for (int j = 0; j < NN; ++j) {
    int dy = yi - j / WWW, dx = xi - j % WWW;
    int d2 = dy * dy + dx * dx;
    if (d2 < bd[TOPK - 1] || (d2 == bd[TOPK - 1] && j < bi[TOPK - 1])) {
      bd[TOPK - 1] = d2; bi[TOPK - 1] = j;
#pragma unroll
      for (int p = TOPK - 1; p > 0; --p) {
        bool sw = (bd[p] < bd[p - 1]) || (bd[p] == bd[p - 1] && bi[p] < bi[p - 1]);
        if (sw) { int td = bd[p]; bd[p] = bd[p-1]; bd[p-1] = td;
                  int ti = bi[p]; bi[p] = bi[p-1]; bi[p-1] = ti; }
      }
    }
  }
#pragma unroll
  for (int t = 0; t < TOPK; ++t) sidx[i * TOPK + t] = bi[t];
}

// ---------------------------------------------------------------------------
// 4) Feature 3-NN fused with the WMMA Gram matrix.  The 16x512 column tile is
//    fetched by the Tensor Data Mover (one DMA per tile, TENSORcnt-tracked).
// ---------------------------------------------------------------------------
__global__ void fknn_kernel(const _Float16* nbf, const float* sq, int* fidx) {
  __shared__ __align__(16) _Float16 sA[32 * CC];   // 32 rows  x 512 K (32 KB)
  __shared__ __align__(16) _Float16 sB[16 * CC];   // 16 cols  x 512 K (16 KB)
  __shared__ float sD[2][16 * 16];                 // per-wave score tile

  const int b  = blockIdx.y;
  const int rb = blockIdx.x * 32;
  const int tid = threadIdx.x;
  const int w = tid >> 5, lane = tid & 31;
  const int m = lane & 15, half = lane >> 4;
  const _Float16* nb = nbf + (size_t)b * NN * CC;
  const float* sqb = sq + (size_t)b * NN;

  // stage 32 rows of A (full K) with async global->LDS copies
  {
    uint32_t dst = lds_off(sA);
    for (int t = tid; t < 32 * CC / 8; t += 64)
      async_copy_b128(dst + t * 16, (const char*)(nb + (size_t)rb * CC) + t * 16);
    wait_asynccnt0();
  }
  __syncthreads();

  const int rowOwned = rb + w * 16 + lane;        // valid for lane < 16
  float sqi = (lane < 16) ? sqb[rowOwned] : 0.f;
  float bd[TOPK]; int bi[TOPK];
#pragma unroll
  for (int t = 0; t < TOPK; ++t) { bd[t] = 3.4e38f; bi[t] = -1; }

  for (int ct = 0; ct < NN / 16; ++ct) {
    const int jb = ct * 16;
    if (w == 0) {   // wave-uniform: one TDM per tile, issued by wave 0
      tdm_load_tile_f16(lds_off(sB), nb + (size_t)jb * CC,
                        /*cols=*/CC, /*rows=*/16, /*stride=*/CC);
      __builtin_amdgcn_s_wait_tensorcnt(0);
    }
    __syncthreads();

    v8f acc;
#pragma unroll
    for (int r = 0; r < 8; ++r) acc[r] = 0.f;
#pragma unroll 4
    for (int kc = 0; kc < CC / 32; ++kc) {
      v16h a = ld_frag(&sA[(w * 16 + m) * CC + kc * 32], half);
      v16h bv = ld_frag(&sB[m * CC + kc * 32], half);
      acc = wmma_f16(a, bv, acc);
    }
    // C layout: vgpr r, lane l -> M = r + 8*half, N = m
#pragma unroll
    for (int r = 0; r < 8; ++r) sD[w][(r + half * 8) * 16 + m] = acc[r];
    __syncthreads();

    if (lane < 16) {
      for (int n = 0; n < 16; ++n) {
        int j = jb + n;
        float d2 = sqi + sqb[j] - 2.f * sD[w][lane * 16 + n];
        if (d2 < bd[TOPK - 1] || (d2 == bd[TOPK - 1] && j < bi[TOPK - 1])) {
          bd[TOPK - 1] = d2; bi[TOPK - 1] = j;
#pragma unroll
          for (int p = TOPK - 1; p > 0; --p) {
            bool sw = (bd[p] < bd[p - 1]) ||
                      (bd[p] == bd[p - 1] && bi[p] < bi[p - 1]);
            if (sw) { float td = bd[p]; bd[p] = bd[p-1]; bd[p-1] = td;
                      int ti = bi[p]; bi[p] = bi[p-1]; bi[p-1] = ti; }
          }
        }
      }
    }
    __syncthreads();
  }
  if (lane < 16) {
#pragma unroll
    for (int t = 0; t < TOPK; ++t)
      fidx[((size_t)b * NN + rowOwned) * TOPK + t] = bi[t];
  }
}

// 5) zero in-degree counters
__global__ void zero_kernel(int* p, int total) {
  int i = blockIdx.x * blockDim.x + threadIdx.x;
  if (i < total) p[i] = 0;
}

// 6) adjacency build (int atomics only; FP order fixed later by sorting)
__global__ void build_adj_kernel(const int* fidx, const int* sidx,
                                 int* cnt, int* srcs) {
  int j = blockIdx.x * blockDim.x + threadIdx.x;
  int b = blockIdx.y;
  if (j >= NN) return;
  int f[TOPK], s[TOPK];
#pragma unroll
  for (int t = 0; t < TOPK; ++t) {
    f[t] = fidx[((size_t)b * NN + j) * TOPK + t];
    s[t] = sidx[j * TOPK + t];
  }
#pragma unroll
  for (int a = 0; a < TOPK; ++a) {
    int tgt = f[a];
    bool hit = (tgt == s[0]) | (tgt == s[1]) | (tgt == s[2]);
    if (hit) {
      int slot = atomicAdd(&cnt[(size_t)b * NN + tgt], 1);
      if (slot < MAXDEG) srcs[((size_t)b * NN + tgt) * MAXDEG + slot] = j;
    }
  }
}

// ---------------------------------------------------------------------------
// 7) WMMA GEMM:  Y = X[N,C](f16) @ W[C,C] + bias, f32 out.  Tile staging via
//    ASYNCcnt-tracked global_load_async_to_lds_b128 (no VGPR round-trip).
// ---------------------------------------------------------------------------
__global__ void gemm_kernel(const _Float16* X, const _Float16* wt4,
                            const float* bq, const float* bk,
                            const float* bv, const float* bs,
                            float* yq, float* yk, float* yv, float* ys) {
  __shared__ __align__(16) _Float16 sA[64 * 32];
  __shared__ __align__(16) _Float16 sB[64 * 32];

  const int z = blockIdx.z;
  const float* bias; float* y;
  if      (z == 0) { bias = bq; y = yq; }
  else if (z == 1) { bias = bk; y = yk; }
  else if (z == 2) { bias = bv; y = yv; }
  else             { bias = bs; y = ys; }
  const _Float16* wt = wt4 + (size_t)z * CC * CC;

  const int rb = blockIdx.x * 64, cb = blockIdx.y * 64;
  const int tid = threadIdx.x;
  const int w = tid >> 5, lane = tid & 31;
  const int m = lane & 15, half = lane >> 4;
  const uint32_t dA = lds_off(sA), dB = lds_off(sB);

  v8f acc[4];
#pragma unroll
  for (int ct = 0; ct < 4; ++ct)
#pragma unroll
    for (int r = 0; r < 8; ++r) acc[ct][r] = 0.f;

  for (int kt = 0; kt < CC / 32; ++kt) {
    const int kb = kt * 32;
    // stage A (64x32) and B = Wt (64x32): 256 x 16B chunks each, async to LDS
    for (int c0 = tid; c0 < 256; c0 += 128) {
      int row = c0 >> 2, q4 = c0 & 3;
      async_copy_b128(dA + c0 * 16,
                      (const char*)(X + (size_t)(rb + row) * CC + kb) + q4 * 16);
      async_copy_b128(dB + c0 * 16,
                      (const char*)(wt + (size_t)(cb + row) * CC + kb) + q4 * 16);
    }
    if (kt + 1 < CC / 32)  // gfx1250 global_prefetch_b8 for the next K tile
      __builtin_prefetch(X + (size_t)(rb + (tid >> 1)) * CC + kb + 32, 0, 1);
    wait_asynccnt0();
    __syncthreads();

    v16h a = ld_frag(&sA[(w * 16 + m) * 32], half);
#pragma unroll
    for (int ct = 0; ct < 4; ++ct) {
      v16h bv16 = ld_frag(&sB[(ct * 16 + m) * 32], half);
      acc[ct] = wmma_f16(a, bv16, acc[ct]);
    }
    __syncthreads();
  }
  // epilogue: M = r + 8*half within the wave strip, N = m within col tile
#pragma unroll
  for (int ct = 0; ct < 4; ++ct) {
    int col = cb + ct * 16 + m;
    float bcol = bias[col];
#pragma unroll
    for (int r = 0; r < 8; ++r) {
      int row = rb + w * 16 + r + half * 8;
      y[(size_t)row * CC + col] = acc[ct][r] + bcol;
    }
  }
}

// ---------------------------------------------------------------------------
// 8) Sparse TransformerConv attention (<=8 sources per target)
// ---------------------------------------------------------------------------
__global__ void attn_kernel(const float* qm, const float* km, const float* vm,
                            const float* basem, const int* cnt,
                            const int* srcs, float* out) {
  __shared__ float red[128];
  __shared__ float alph[MAXDEG];
  __shared__ int   slist[MAXDEG];
  __shared__ int   sdeg;
  const int i = blockIdx.x, tid = threadIdx.x;
  const float scale = 0.044194173824159216f;  // 1/sqrt(512)

  if (tid == 0) {
    int d = cnt[i]; if (d > MAXDEG) d = MAXDEG;
    int tmp[MAXDEG];
    for (int t = 0; t < d; ++t) tmp[t] = srcs[(size_t)i * MAXDEG + t];
    for (int a = 1; a < d; ++a) {            // insertion sort (deterministic)
      int v = tmp[a], p = a - 1;
      while (p >= 0 && tmp[p] > v) { tmp[p + 1] = tmp[p]; --p; }
      tmp[p + 1] = v;
    }
    for (int t = 0; t < d; ++t) slist[t] = tmp[t];
    sdeg = d;
  }
  __syncthreads();
  const int deg = sdeg;

  for (int j = 0; j < deg; ++j) {
    const int sj = slist[j];
    float p = 0.f;
    for (int c = tid; c < CC; c += 128)
      p += qm[(size_t)i * CC + c] * km[(size_t)sj * CC + c];
    red[tid] = p;
    __syncthreads();
    for (int st = 64; st > 0; st >>= 1) {
      if (tid < st) red[tid] += red[tid + st];
      __syncthreads();
    }
    if (tid == 0) alph[j] = red[0] * scale;
    __syncthreads();
  }
  if (tid == 0 && deg > 0) {
    float mx = alph[0];
    for (int j = 1; j < deg; ++j) mx = fmaxf(mx, alph[j]);
    float sum = 0.f;
    for (int j = 0; j < deg; ++j) { float e = expf(alph[j] - mx); alph[j] = e; sum += e; }
    float inv = 1.f / sum;
    for (int j = 0; j < deg; ++j) alph[j] *= inv;
  }
  __syncthreads();
  for (int c = tid; c < CC; c += 128) {
    float a = basem[(size_t)i * CC + c];
    for (int j = 0; j < deg; ++j)
      a += alph[j] * vm[(size_t)slist[j] * CC + c];
    out[(size_t)i * CC + c] = a;
  }
}

// 9) ReLU + LayerNorm over C, emit f16 for the next layer's WMMA input
__global__ void relu_ln_kernel(const float* h, const float* g, const float* bb,
                               _Float16* out) {
  __shared__ float r1[128], r2[128];
  const int i = blockIdx.x, tid = threadIdx.x;
  float vals[CC / 128];
  float s = 0.f, s2 = 0.f;
#pragma unroll
  for (int t = 0; t < CC / 128; ++t) {
    float v = h[(size_t)i * CC + tid + t * 128];
    v = fmaxf(v, 0.f);
    vals[t] = v; s += v; s2 += v * v;
  }
  r1[tid] = s; r2[tid] = s2;
  __syncthreads();
  for (int st = 64; st > 0; st >>= 1) {
    if (tid < st) { r1[tid] += r1[tid + st]; r2[tid] += r2[tid + st]; }
    __syncthreads();
  }
  float mean = r1[0] * (1.f / CC);
  float var  = fmaxf(r2[0] * (1.f / CC) - mean * mean, 0.f);
  float rstd = rsqrtf(var + LN_EPS);
#pragma unroll
  for (int t = 0; t < CC / 128; ++t) {
    int c = tid + t * 128;
    out[(size_t)i * CC + c] = (_Float16)((vals[t] - mean) * rstd * g[c] + bb[c]);
  }
}

// 10) final: out = sigmoid(h2) * x3
__global__ void final_kernel(const float* h2, const float* x3, float* out, int total) {
  int i = blockIdx.x * blockDim.x + threadIdx.x;
  if (i < total) {
    float v = h2[i];
    out[i] = x3[i] / (1.f + expf(-v));
  }
}

// ---------------------------------------------------------------------------
extern "C" void kernel_launch(void* const* d_in, const int* in_sizes, int n_in,
                              void* d_out, int out_size, void* d_ws, size_t ws_size,
                              hipStream_t stream) {
  (void)in_sizes; (void)n_in; (void)out_size; (void)ws_size;
  const float* x3 = (const float*)d_in[0];
  const float* W[8]  = { (const float*)d_in[1],  (const float*)d_in[3],
                         (const float*)d_in[5],  (const float*)d_in[7],
                         (const float*)d_in[9],  (const float*)d_in[11],
                         (const float*)d_in[13], (const float*)d_in[15] };
  const float* Bq1 = (const float*)d_in[2];  const float* Bk1 = (const float*)d_in[4];
  const float* Bv1 = (const float*)d_in[6];  const float* Bs1 = (const float*)d_in[8];
  const float* Bq2 = (const float*)d_in[10]; const float* Bk2 = (const float*)d_in[12];
  const float* Bv2 = (const float*)d_in[14]; const float* Bs2 = (const float*)d_in[16];
  const float* lng = (const float*)d_in[17]; const float* lnb = (const float*)d_in[18];
  float* outp = (float*)d_out;

  // --- workspace carve-out (256B aligned) ---
  char* wsb = (char*)d_ws;
  size_t off = 0;
  auto alloc = [&](size_t bytes) -> char* {
    char* p = wsb + off;
    off += (bytes + 255) & ~(size_t)255;
    return p;
  };
  _Float16* wtBf    = (_Float16*)alloc((size_t)8 * CC * CC * 2);
  _Float16* nodesBf = (_Float16*)alloc((size_t)BB * NN * CC * 2);
  float*    sq      = (float*)   alloc((size_t)BB * NN * 4);
  int*      sidx    = (int*)     alloc((size_t)NN * TOPK * 4);
  int*      fidx    = (int*)     alloc((size_t)BB * NN * TOPK * 4);
  int*      cnt     = (int*)     alloc((size_t)BB * NN * 4);
  int*      srcs    = (int*)     alloc((size_t)BB * NN * MAXDEG * 4);
  float*    qb      = (float*)   alloc((size_t)NN * CC * 4);
  float*    kb      = (float*)   alloc((size_t)NN * CC * 4);
  float*    vb      = (float*)   alloc((size_t)NN * CC * 4);
  float*    baseb   = (float*)   alloc((size_t)NN * CC * 4);
  float*    h1      = (float*)   alloc((size_t)NN * CC * 4);
  _Float16* h1bf    = (_Float16*)alloc((size_t)NN * CC * 2);
  float*    h2      = (float*)   alloc((size_t)NN * CC * 4);

  // --- shared preprocessing (whole batch) ---
  cvt_weights_kernel<<<dim3(CC * CC / 256, 8), 256, 0, stream>>>(
      W[0], W[1], W[2], W[3], W[4], W[5], W[6], W[7], wtBf);

  const int totNodes = BB * NN * CC;
  cvt_f16_kernel<<<(totNodes + 255) / 256, 256, 0, stream>>>(x3, nodesBf, totNodes);
  sqnorm_kernel<<<BB * NN, 128, 0, stream>>>(x3, sq);
  spatial_knn_kernel<<<(NN + 255) / 256, 256, 0, stream>>>(sidx);
  fknn_kernel<<<dim3(NN / 32, BB), 64, 0, stream>>>(nodesBf, sq, fidx);
  zero_kernel<<<(BB * NN + 255) / 256, 256, 0, stream>>>(cnt, BB * NN);
  build_adj_kernel<<<dim3((NN + 255) / 256, BB), 256, 0, stream>>>(fidx, sidx, cnt, srcs);

  // --- per sample ---
  for (int b = 0; b < BB; ++b) {
    const _Float16* nb = nodesBf + (size_t)b * NN * CC;
    const float*    xb = x3 + (size_t)b * NN * CC;
    const int* cb = cnt + (size_t)b * NN;
    const int* sb = srcs + (size_t)b * NN * MAXDEG;

    // layer 1: q/k/v/skip in one fused launch (blockIdx.z selects matrix)
    gemm_kernel<<<dim3(NN / 64, CC / 64, 4), 128, 0, stream>>>(
        nb, wtBf, Bq1, Bk1, Bv1, Bs1, qb, kb, vb, baseb);
    attn_kernel<<<NN, 128, 0, stream>>>(qb, kb, vb, baseb, cb, sb, h1);
    relu_ln_kernel<<<NN, 128, 0, stream>>>(h1, lng, lnb, h1bf);

    // layer 2
    gemm_kernel<<<dim3(NN / 64, CC / 64, 4), 128, 0, stream>>>(
        h1bf, wtBf + (size_t)4 * CC * CC, Bq2, Bk2, Bv2, Bs2, qb, kb, vb, baseb);
    attn_kernel<<<NN, 128, 0, stream>>>(qb, kb, vb, baseb, cb, sb, h2);

    final_kernel<<<(NN * CC + 255) / 256, 256, 0, stream>>>(
        h2, xb, outp + (size_t)b * NN * CC, NN * CC);
  }
}
